// SequenceStateSpace_58841051955476
// MI455X (gfx1250) — matrix operand order repaired and checked by reading, not compiled, not run on hardware
//
#include <hip/hip_runtime.h>

// ---------------------------------------------------------------------------
// Fused GEMM(+bias,+sigmoid) x2 -> gated linear scan for MI455X (gfx1250).
// B=8, T=4096, D=1024.  delta = sigmoid(seq@Wd^T + bd), value = seq@Wv^T + bv,
// state_t = (1-delta)*state_{t-1} + delta*value  (scan over T).
// Compute-bound (~500 flop/byte) -> bf16 WMMA 16x16x32 with f32 accumulate.
// Each wave owns 2 M-subtiles so every B-fragment feeds 2 WMMAs (1.0 frag
// loads per WMMA).  out is streamed with NT stores to preserve L2 for seq/W;
// next seq chunk is prefetched during the K-loop.
// ---------------------------------------------------------------------------

typedef __attribute__((ext_vector_type(16))) __bf16 v16bf;
typedef __attribute__((ext_vector_type(8)))  __bf16 v8bf;
typedef __attribute__((ext_vector_type(8)))  float  v8f;

#define B_DIM 8
#define T_DIM 4096
#define D_DIM 1024
#define MCH   128             // T-rows per chunk (8 M-subtiles of 16)
#define NCOL  32              // output columns per workgroup (2 N-subtiles)
#define NTHREADS 256          // 8 wave32 waves
#define LSTR (D_DIM + 8)      // LDS A row stride in bf16 elems (bank-conflict pad)

__device__ __forceinline__ unsigned short f2bf(float f) {
  unsigned int u = __builtin_bit_cast(unsigned int, f);
  u += 0x7FFFu + ((u >> 16) & 1u);          // round-to-nearest-even
  return (unsigned short)(u >> 16);
}

__device__ __forceinline__ float fast_sigmoid(float x) {
  // v_exp_f32 + v_rcp_f32 instead of the IEEE v_div_scale chain
  return __builtin_amdgcn_rcpf(1.0f + __expf(-x));
}

__device__ __forceinline__ v8f wmma_bf16(v16bf a, v16bf b, v8f c) {
  // D(f32 16x16) = A(bf16 16x32) * B(bf16 32x16) + C
  return __builtin_amdgcn_wmma_f32_16x16x32_bf16(
      /*neg_a=*/false, a, /*neg_b=*/false, b,
      /*c_mod=*/(short)0, c, /*reuse_a=*/false, /*reuse_b=*/false);
}

// --- weight fp32 -> bf16 conversion (runs once per launch, 2x 4MiB) --------
__global__ void cvt_w_bf16(const float* __restrict__ src,
                           unsigned short* __restrict__ dst, int n) {
  int i = (blockIdx.x * blockDim.x + threadIdx.x) * 4;
  if (i + 3 < n) {
    float4 f = *(const float4*)(src + i);
    uint2 p;
    p.x = (unsigned int)f2bf(f.x) | ((unsigned int)f2bf(f.y) << 16);
    p.y = (unsigned int)f2bf(f.z) | ((unsigned int)f2bf(f.w) << 16);
    *(uint2*)(dst + i) = p;
  }
}

// --- fused persistent kernel ----------------------------------------------
// grid = (D/NCOL, B); each WG owns (batch b, columns e0..e0+31) for all T.
__global__ void __launch_bounds__(NTHREADS, 1)
fused_ssm(const float* __restrict__ seq,
          const unsigned short* __restrict__ Wd,   // bf16 [D][D], row = e
          const unsigned short* __restrict__ Wv,   // bf16 [D][D]
          const float* __restrict__ bdelta,
          const float* __restrict__ bvalue,
          float* __restrict__ out) {
  extern __shared__ char smem[];
  unsigned short* lds_a = (unsigned short*)smem;                       // [MCH][LSTR] bf16
  float2* dvbuf = (float2*)(smem + (size_t)MCH * LSTR * sizeof(unsigned short)); // [MCH][NCOL]

  const int tid    = threadIdx.x;
  const int lane   = tid & 31;
  const int wave   = tid >> 5;
  const int lane16 = lane & 15;
  const int khalf  = lane >> 4;      // A: +8 K offset half; B: +8 K offset half
  const int mh     = wave & 3;       // 4 M-pairs: rows [mh*32, mh*32+31]
  const int nsub   = wave >> 2;      // 2 N-subtiles of 16 cols

  const int b  = blockIdx.y;
  const int e0 = blockIdx.x * NCOL;

  // B-matrix lane addressing: lane holds column N = e (row of W), K split {0..7,16..23}/{8..15,24..31}.
  const int colW = e0 + nsub * 16 + lane16;
  const float bd_r = bdelta[colW];
  const float bv_r = bvalue[colW];
  const unsigned short* WdRow = Wd + (size_t)colW * D_DIM;
  const unsigned short* WvRow = Wv + (size_t)colW * D_DIM;

  // A-matrix lane addressing: lane holds row M, K split {0..7,16..23}/{8..15,24..31}.
  const int arow0 = mh * 32 + lane16;        // first M-subtile
  const int arow1 = arow0 + 16;              // second M-subtile

  float st = 0.0f;  // scan state (live only in lanes tid < NCOL)

  for (int t0 = 0; t0 < T_DIM; t0 += MCH) {
    // ---- stage seq chunk [MCH][D] fp32 -> LDS bf16 (read HBM exactly once)
    const float* srcRow = seq + ((size_t)b * T_DIM + t0) * D_DIM;
    #pragma unroll 4
    for (int i = 0; i < MCH; ++i) {
      float4 f = *(const float4*)(srcRow + (size_t)i * D_DIM + tid * 4);
      uint2 p;
      p.x = (unsigned int)f2bf(f.x) | ((unsigned int)f2bf(f.y) << 16);
      p.y = (unsigned int)f2bf(f.z) | ((unsigned int)f2bf(f.w) << 16);
      *(uint2*)(lds_a + (size_t)i * LSTR + tid * 4) = p;
    }
    __syncthreads();

    // ---- prefetch next chunk into L2 while the K-loop runs (counter-free)
    if (t0 + MCH < T_DIM) {
      const char* nxt = (const char*)(srcRow + (size_t)MCH * D_DIM);
      #pragma unroll
      for (int p = 0; p < 16; ++p)   // 256 thr * 16 * 128B = 512 KiB chunk
        __builtin_prefetch(nxt + ((size_t)p * NTHREADS + tid) * 128, 0, 3);
    }

    // ---- dual GEMM over K=1024: two 16x16 tiles x {delta,value} per wave
    v8f accd0 = {}, accd1 = {};
    v8f accv0 = {}, accv1 = {};
    const unsigned short* aBase0 = lds_a + (size_t)arow0 * LSTR;
    const unsigned short* aBase1 = lds_a + (size_t)arow1 * LSTR;
    #pragma unroll 2
    for (int k0 = 0; k0 < D_DIM; k0 += 32) {
      const int ka = k0 + khalf * 8;
      v16bf a0, a1, bdm, bvm;
      ((v8bf*)&a0)[0] = *(const v8bf*)(aBase0 + ka);        // K = ka..ka+7
      ((v8bf*)&a0)[1] = *(const v8bf*)(aBase0 + ka + 16);   // K = ka+16..ka+23
      ((v8bf*)&a1)[0] = *(const v8bf*)(aBase1 + ka);
      ((v8bf*)&a1)[1] = *(const v8bf*)(aBase1 + ka + 16);
      const int kb = k0 + khalf * 8;
      ((v8bf*)&bdm)[0] = *(const v8bf*)(WdRow + kb);
      ((v8bf*)&bdm)[1] = *(const v8bf*)(WdRow + kb + 16);
      ((v8bf*)&bvm)[0] = *(const v8bf*)(WvRow + kb);
      ((v8bf*)&bvm)[1] = *(const v8bf*)(WvRow + kb + 16);
      accd0 = wmma_bf16(a0, bdm, accd0);   // each B fragment feeds 2 WMMAs
      accd1 = wmma_bf16(a1, bdm, accd1);
      accv0 = wmma_bf16(a0, bvm, accv0);
      accv1 = wmma_bf16(a1, bvm, accv1);
    }

    // ---- epilogue: bias + sigmoid, deposit (delta,value) pairs in LDS
    {
      const int col = nsub * 16 + lane16;
      #pragma unroll
      for (int m = 0; m < 2; ++m) {
        const int rowbase = mh * 32 + m * 16 + khalf * 8;  // C/D: VGPR v -> M=base+v
        v8f ad = m ? accd1 : accd0;
        v8f av = m ? accv1 : accv0;
        #pragma unroll
        for (int v = 0; v < 8; ++v) {
          float d   = fast_sigmoid(ad[v] + bd_r);
          float val = av[v] + bv_r;
          dvbuf[(rowbase + v) * NCOL + col] = make_float2(d, val);
        }
      }
    }
    __syncthreads();

    // ---- gated linear scan: wave 0, one lane per column, MCH sequential steps
    if (tid < NCOL) {
      float s = st;
      float* op = out + ((size_t)b * T_DIM + t0) * D_DIM + e0 + tid;
      #pragma unroll 4
      for (int r = 0; r < MCH; ++r) {
        float2 p = dvbuf[r * NCOL + tid];
        s += p.x * (p.y - s);                       // (1-d)s + d v
        __builtin_nontemporal_store(s, op + (size_t)r * D_DIM);  // NT: spare L2
      }
      st = s;
    }
    // next chunk's lds_a staging is disjoint from dvbuf; the post-stage
    // __syncthreads() already orders scan reads vs. the next dvbuf writes.
  }
}

extern "C" void kernel_launch(void* const* d_in, const int* in_sizes, int n_in,
                              void* d_out, int out_size, void* d_ws, size_t ws_size,
                              hipStream_t stream) {
  const float* seq = (const float*)d_in[0];
  const float* Wd  = (const float*)d_in[1];
  const float* bd  = (const float*)d_in[2];
  const float* Wv  = (const float*)d_in[3];
  const float* bv  = (const float*)d_in[4];
  float* out = (float*)d_out;

  unsigned short* Wdb = (unsigned short*)d_ws;                 // bf16 weights
  unsigned short* Wvb = Wdb + (size_t)D_DIM * D_DIM;           // needs 4 MiB ws

  const int nW = D_DIM * D_DIM;
  cvt_w_bf16<<<nW / (4 * 256), 256, 0, stream>>>(Wd, Wdb, nW);
  cvt_w_bf16<<<nW / (4 * 256), 256, 0, stream>>>(Wv, Wvb, nW);

  dim3 grid(D_DIM / NCOL, B_DIM);  // 32 x 8 = 256 workgroups
  size_t shmem = (size_t)MCH * LSTR * sizeof(unsigned short)   // 264192 B
               + (size_t)MCH * NCOL * sizeof(float2);          // +32768 B
  fused_ssm<<<grid, NTHREADS, shmem, stream>>>(seq, Wdb, Wvb, bd, bv, out);
}
